// FlashAttentionWrapper_5634997092675
// MI455X (gfx1250) — matrix operand order repaired and checked by reading, not compiled, not run on hardware
//
#include <hip/hip_runtime.h>

// ---------------------------------------------------------------------------
// Types for CDNA5 WMMA (gfx1250, wave32)
// ---------------------------------------------------------------------------
typedef __attribute__((ext_vector_type(16))) __bf16  bf16x16;
typedef __attribute__((ext_vector_type(8)))  float   floatx8;
typedef __attribute__((ext_vector_type(4)))  unsigned int u32x4;

union FragBF {
    bf16x16 v;
    u32x4   q[2];
};

__device__ __forceinline__ floatx8 zero8() {
    floatx8 z = {0.f, 0.f, 0.f, 0.f, 0.f, 0.f, 0.f, 0.f};
    return z;
}

__device__ __forceinline__ floatx8 wmma_bf16(bf16x16 a, bf16x16 b, floatx8 c) {
    // D = A(16x32 bf16) x B(32x16 bf16) + C(16x16 f32)
    return __builtin_amdgcn_wmma_f32_16x16x32_bf16(
        /*neg_a=*/false, a, /*neg_b=*/false, b,
        /*c_mod=*/(short)0, c, /*reuse_a=*/false, /*reuse_b=*/false);
}

// A-fragment (16x32, M-major, K contiguous in memory):
//   lane L: M = L&15 ; halves h: K = (h<8?0:16) + (L>=16?8:0) + (h&7)
__device__ __forceinline__ bf16x16 load_afrag(const __bf16* base, int ld, int lane) {
    FragBF f;
    const int row  = lane & 15;
    const int koff = (lane >> 4) << 3;           // 0 or 8
    const __bf16* p = base + row * ld + koff;
    f.q[0] = *(const u32x4*)(p);
    f.q[1] = *(const u32x4*)(p + 16);
    return f.v;
}

// B-fragment (32x16, stored [N][K], K contiguous):
//   lane L: N = L&15 ; halves h: K = (h<8?0:16) + (L>=16?8:0) + (h&7)
__device__ __forceinline__ bf16x16 load_bfrag(const __bf16* base, int ld, int lane) {
    FragBF f;
    const int col  = lane & 15;
    const int koff = (lane >> 4) << 3;           // 0 or 8
    const __bf16* p = base + col * ld + koff;
    f.q[0] = *(const u32x4*)(p);
    f.q[1] = *(const u32x4*)(p + 16);
    return f.v;
}

// CDNA5 async copy: 16B per lane, global -> LDS, tracked by ASYNCcnt.
// VDST holds the per-lane LDS byte offset; VADDR holds the 64-bit global addr.
__device__ __forceinline__ void async_g2l_b128(void* lds_ptr, const void* gptr) {
    unsigned ldsoff = (unsigned)(unsigned long long)lds_ptr;   // addr[31:0] = LDS offset
    asm volatile("global_load_async_to_lds_b128 %0, %1, off"
                 :: "v"(ldsoff), "v"(gptr) : "memory");
}

__device__ __forceinline__ void wait_asynccnt0() {
    asm volatile("s_wait_asynccnt 0x0" ::: "memory");
}

// ---------------------------------------------------------------------------
// fp32 -> bf16 convert
// ---------------------------------------------------------------------------
__global__ __launch_bounds__(256) void cvt_f32_bf16(const float* __restrict__ in,
                                                    __bf16* __restrict__ out, int n) {
    int i = blockIdx.x * 256 + threadIdx.x;
    if (i < n) out[i] = (__bf16)in[i];
}

// ---------------------------------------------------------------------------
// bf16 NT-GEMM: C[m,n] = sum_k A[m,k] * B[n,k]   (both K-contiguous)
// Block: 256 threads (8 waves), tile 128(M) x 64(N), staged K slab = 64.
// MODE 0: store f32 row-major into Cf (ld = N)
// MODE 1: QKV scatter epilogue -> Q(B,H,S,64) K(B,H,S,64) V^T(B,H,64,S) bf16
// ---------------------------------------------------------------------------
template <int MODE>
__global__ __launch_bounds__(256) void gemm_bf16_nt(
    const __bf16* __restrict__ A, const __bf16* __restrict__ Bw,
    float* __restrict__ Cf,
    __bf16* __restrict__ Qo, __bf16* __restrict__ Ko, __bf16* __restrict__ Vo,
    int M, int N, int K, int S)
{
    __shared__ alignas(16) __bf16 As[128][72];   // 64 cols + 8 pad
    __shared__ alignas(16) __bf16 Bs[64][72];

    const int lane = threadIdx.x & 31;
    const int wave = threadIdx.x >> 5;
    const int wm = wave >> 1;                    // 0..3
    const int wn = wave & 1;                     // 0..1
    const int m0 = blockIdx.y * 128;
    const int n0 = blockIdx.x * 64;
    const int hi  = lane >> 4;
    const int col = lane & 15;

    floatx8 acc[2][2];
    acc[0][0] = zero8(); acc[0][1] = zero8();
    acc[1][0] = zero8(); acc[1][1] = zero8();

    for (int kk = 0; kk < K; kk += 64) {
        __syncthreads();
        // stage A slab: 128 rows x 64 cols = 1024 x 16B chunks, 4 per thread
#pragma unroll
        for (int i = 0; i < 4; ++i) {
            int idx = threadIdx.x + i * 256;
            int r = idx >> 3, c = (idx & 7) << 3;
            *(u32x4*)&As[r][c] = *(const u32x4*)(A + (long)(m0 + r) * K + kk + c);
        }
        // stage B slab: 64 rows x 64 cols = 512 chunks, 2 per thread
#pragma unroll
        for (int i = 0; i < 2; ++i) {
            int idx = threadIdx.x + i * 256;
            int r = idx >> 3, c = (idx & 7) << 3;
            *(u32x4*)&Bs[r][c] = *(const u32x4*)(Bw + (long)(n0 + r) * K + kk + c);
        }
        // prefetch next K slab into cache while we compute this one
        if (kk + 64 < K) {
            __builtin_prefetch(A  + (long)(m0 + (threadIdx.x & 127)) * K + kk + 64, 0, 0);
            __builtin_prefetch(Bw + (long)(n0 + (threadIdx.x & 63))  * K + kk + 64, 0, 0);
        }
        __syncthreads();

#pragma unroll
        for (int kh = 0; kh < 2; ++kh) {
            bf16x16 a0 = load_afrag(&As[wm * 32][kh * 32],      72, lane);
            bf16x16 a1 = load_afrag(&As[wm * 32 + 16][kh * 32], 72, lane);
            bf16x16 b0 = load_bfrag(&Bs[wn * 32][kh * 32],      72, lane);
            bf16x16 b1 = load_bfrag(&Bs[wn * 32 + 16][kh * 32], 72, lane);

            acc[0][0] = wmma_bf16(a0, b0, acc[0][0]);
            acc[0][1] = wmma_bf16(a0, b1, acc[0][1]);
            acc[1][0] = wmma_bf16(a1, b0, acc[1][0]);
            acc[1][1] = wmma_bf16(a1, b1, acc[1][1]);
        }
    }

#pragma unroll
    for (int i = 0; i < 2; ++i) {
#pragma unroll
        for (int j = 0; j < 2; ++j) {
#pragma unroll
            for (int r = 0; r < 8; ++r) {
                const int row = m0 + wm * 32 + i * 16 + r + hi * 8;   // C layout: M = r + 8*hi
                const int cc  = n0 + wn * 32 + j * 16 + col;          //           N = lane&15
                const float v = acc[i][j][r];
                if (MODE == 0) {
                    Cf[(long)row * N + cc] = v;
                } else {
                    const int b = row / S, s = row % S;
                    int e = cc;
                    if (e < 1024) {                     // Q: (B,H,S,64)
                        int h = e >> 6, d = e & 63;
                        Qo[(((long)b * 16 + h) * S + s) * 64 + d] = (__bf16)v;
                    } else if (e < 2048) {              // K: (B,H,S,64)
                        e -= 1024;
                        int h = e >> 6, d = e & 63;
                        Ko[(((long)b * 16 + h) * S + s) * 64 + d] = (__bf16)v;
                    } else {                            // V^T: (B,H,64,S)
                        e -= 2048;
                        int h = e >> 6, d = e & 63;
                        Vo[(((long)b * 16 + h) * 64 + d) * (long)S + s] = (__bf16)v;
                    }
                }
            }
        }
    }
}

// ---------------------------------------------------------------------------
// Flash attention (online softmax).
// Grid: (S/64, B*H). Block: 128 threads = 4 waves; wave owns 16 query rows.
// K tile: 32 keys x 64 d ; V^T tile: 64 d x 32 keys ; staged with CDNA5
// async global->LDS copies (ASYNCcnt) then consumed by WMMA.
// ---------------------------------------------------------------------------
__global__ __launch_bounds__(128) void flash_attn_kernel(
    const __bf16* __restrict__ Q, const __bf16* __restrict__ Kb,
    const __bf16* __restrict__ VT, __bf16* __restrict__ O,
    const int* __restrict__ causal_p, int S, int H)
{
    __shared__ alignas(16) __bf16 Ks[32][72];       // [k][d], pad 8
    __shared__ alignas(16) __bf16 Vs[64][40];       // [d][k], pad 8
    __shared__ alignas(16) __bf16 Ps[4][16][40];    // per-wave P staging [q][k]

    const float SCALE = 0.125f;                     // 1/sqrt(64)
    const float NEG_BIG = -1.0e30f;

    const int lane = threadIdx.x & 31;
    const int wave = threadIdx.x >> 5;
    const int bh = blockIdx.y;
    const int qb = blockIdx.x * 64;
    const int q0w = qb + wave * 16;
    const int hi  = lane >> 4;
    const int col = lane & 15;
    const int causal = *causal_p;

    // Q fragments for this wave: 16 rows x 64 d -> two A-fragments (K=32 each)
    const __bf16* qbase = Q + ((long)bh * S + q0w) * 64;
    const bf16x16 qa0 = load_afrag(qbase, 64, lane);
    const bf16x16 qa1 = load_afrag(qbase + 32, 64, lane);

    float m[8], l[8];
    floatx8 acc[4];
#pragma unroll
    for (int r = 0; r < 8; ++r) { m[r] = NEG_BIG; l[r] = 0.f; }
#pragma unroll
    for (int dn = 0; dn < 4; ++dn) acc[dn] = zero8();

    const int nkt = causal ? (qb + 64) / 32 : (S / 32);

    for (int kt = 0; kt < nkt; ++kt) {
        const int k0 = kt * 32;
        __syncthreads();   // previous iteration's consumers done before restage
        // async stage K tile (32 x 64 bf16) and V^T tile (64 x 32 bf16)
        for (int i = threadIdx.x; i < 256; i += 128) {
            int r = i >> 3, c = (i & 7) << 3;
            async_g2l_b128(&Ks[r][c], Kb + ((long)bh * S + k0 + r) * 64 + c);
        }
        for (int i = threadIdx.x; i < 256; i += 128) {
            int r = i >> 2, c = (i & 3) << 3;
            async_g2l_b128(&Vs[r][c], VT + ((long)bh * 64 + r) * (long)S + k0 + c);
        }
        wait_asynccnt0();  // my wave's async copies landed in LDS
        __syncthreads();   // everyone's copies landed

        // scores: S = Q(16x64) * K^T -> two 16x16 tiles (k 0..15, 16..31)
        floatx8 s0 = zero8(), s1 = zero8();
        s0 = wmma_bf16(qa0, load_bfrag(&Ks[0][0],  72, lane), s0);
        s0 = wmma_bf16(qa1, load_bfrag(&Ks[0][32], 72, lane), s0);
        s1 = wmma_bf16(qa0, load_bfrag(&Ks[16][0],  72, lane), s1);
        s1 = wmma_bf16(qa1, load_bfrag(&Ks[16][32], 72, lane), s1);

        float p0[8], p1[8], mx[8], rs[8];
#pragma unroll
        for (int r = 0; r < 8; ++r) {
            float v0 = s0[r] * SCALE;
            float v1 = s1[r] * SCALE;
            if (causal) {
                const int qg = q0w + r + hi * 8;
                if (k0 + col > qg)      v0 = NEG_BIG;
                if (k0 + 16 + col > qg) v1 = NEG_BIG;
            }
            p0[r] = v0; p1[r] = v1;
            mx[r] = fmaxf(v0, v1);
        }
        // row max: reduce across the 16 lanes of this half-wave
#pragma unroll
        for (int r = 0; r < 8; ++r) {
#pragma unroll
            for (int off = 1; off < 16; off <<= 1)
                mx[r] = fmaxf(mx[r], __shfl_xor(mx[r], off, 32));
        }
#pragma unroll
        for (int r = 0; r < 8; ++r) {
            const float mnew  = fmaxf(m[r], mx[r]);
            const float alpha = __expf(m[r] - mnew);
            m[r] = mnew;
            const float e0 = __expf(p0[r] - mnew);
            const float e1 = __expf(p1[r] - mnew);
            p0[r] = e0; p1[r] = e1;
            rs[r] = e0 + e1;
            l[r] *= alpha;
#pragma unroll
            for (int dn = 0; dn < 4; ++dn) acc[dn][r] = acc[dn][r] * alpha;
        }
#pragma unroll
        for (int r = 0; r < 8; ++r) {
#pragma unroll
            for (int off = 1; off < 16; off <<= 1)
                rs[r] += __shfl_xor(rs[r], off, 32);
            l[r] += rs[r];
        }

        // C-layout -> A-layout for P via per-wave LDS staging
#pragma unroll
        for (int r = 0; r < 8; ++r) {
            const int row = r + hi * 8;
            Ps[wave][row][col]      = (__bf16)p0[r];
            Ps[wave][row][16 + col] = (__bf16)p1[r];
        }
        const bf16x16 pa = load_afrag(&Ps[wave][0][0], 40, lane);
#pragma unroll
        for (int dn = 0; dn < 4; ++dn) {
            const bf16x16 vb = load_bfrag(&Vs[dn * 16][0], 40, lane);
            acc[dn] = wmma_bf16(pa, vb, acc[dn]);
        }
    }

    // normalize and store O in (B, S, H*64) order (ready for out-projection)
    const int b = bh / H, h = bh % H;
#pragma unroll
    for (int r = 0; r < 8; ++r) {
        const float inv = 1.0f / l[r];
        const int qg = q0w + r + hi * 8;
        __bf16* obase = O + ((long)b * S + qg) * (long)(64 * H) + h * 64;
#pragma unroll
        for (int dn = 0; dn < 4; ++dn)
            obase[dn * 16 + col] = (__bf16)(acc[dn][r] * inv);
    }
}

// ---------------------------------------------------------------------------
// Host launcher
// ---------------------------------------------------------------------------
extern "C" void kernel_launch(void* const* d_in, const int* in_sizes, int n_in,
                              void* d_out, int out_size, void* d_ws, size_t ws_size,
                              hipStream_t stream) {
    const float* x      = (const float*)d_in[0];   // (2, 2048, 1024)
    const float* w_qkv  = (const float*)d_in[1];   // (3072, 1024)
    const float* w_out  = (const float*)d_in[2];   // (1024, 1024)
    const int*   is_cau = (const int*)d_in[3];
    float* out = (float*)d_out;                    // (2, 2048, 1024)

    const int B = 2, S = 2048, D = 1024, H = 16;
    const int NX  = B * S * D;     // 4194304
    const int NWQ = 3 * D * D;     // 3145728
    const int NWO = D * D;         // 1048576

    __bf16* ws      = (__bf16*)d_ws;
    __bf16* xbf     = ws;
    __bf16* wqkvbf  = xbf + NX;
    __bf16* woutbf  = wqkvbf + NWQ;
    __bf16* Qbf     = woutbf + NWO;
    __bf16* Kbf     = Qbf + NX;
    __bf16* VTbf    = Kbf + NX;
    __bf16* Obf     = VTbf + NX;
    (void)in_sizes; (void)n_in; (void)out_size; (void)ws_size;

    cvt_f32_bf16<<<(NX  + 255) / 256, 256, 0, stream>>>(x,     xbf,    NX);
    cvt_f32_bf16<<<(NWQ + 255) / 256, 256, 0, stream>>>(w_qkv, wqkvbf, NWQ);
    cvt_f32_bf16<<<(NWO + 255) / 256, 256, 0, stream>>>(w_out, woutbf, NWO);

    // QKV projection with Q/K/V^T scatter epilogue
    dim3 g1(3 * D / 64, B * S / 128);
    gemm_bf16_nt<1><<<g1, 256, 0, stream>>>(xbf, wqkvbf, nullptr,
                                            Qbf, Kbf, VTbf,
                                            B * S, 3 * D, D, S);

    // flash attention
    dim3 g2(S / 64, B * H);
    flash_attn_kernel<<<g2, 128, 0, stream>>>(Qbf, Kbf, VTbf, Obf, is_cau, S, H);

    // output projection -> fp32
    dim3 g3(D / 64, B * S / 128);
    gemm_bf16_nt<0><<<g3, 256, 0, stream>>>(Obf, woutbf, out,
                                            nullptr, nullptr, nullptr,
                                            B * S, D, D, S);
}
